// OverlapWindowAttention_21852793602095
// MI455X (gfx1250) — matrix operand order repaired and checked by reading, not compiled, not run on hardware
//
#include <hip/hip_runtime.h>
#include <hip/hip_bf16.h>
#include <stdint.h>

#define WSZ    16
#define OWSZ   24
#define HEADS  6
#define DIM    192
#define HDIM   32
#define NQ     256
#define NKV    576
#define BATCH  2
#define IMGH   160
#define IMGW   160
#define PADH   168
#define NWIN1  10            // windows per side
#define NWIN   100           // windows per image
#define NTOK   (BATCH*IMGH*IMGW)   // 51200
#define NPAD   (BATCH*PADH*PADH)   // 56448
#define QSCALE 0.17677669529663687f

typedef __attribute__((ext_vector_type(16))) __bf16 v16bf;
typedef __attribute__((ext_vector_type(8)))  float  v8f;

union Frag { v16bf v; unsigned u[8]; };

// native bf16 converts (let clang emit v_cvt_*bf16* instead of sw rounding)
__device__ __forceinline__ unsigned short f2bf(float f) {
  union { __bf16 h; unsigned short u; } v; v.h = (__bf16)f; return v.u;
}
__device__ __forceinline__ float bf2f(unsigned short b) {
  union { unsigned u; float f; } v; v.u = ((unsigned)b) << 16; return v.f;
}
__device__ __forceinline__ unsigned packbf2(float a, float b) {
  union { __bf16 h[2]; unsigned u; } v;
  v.h[0] = (__bf16)a; v.h[1] = (__bf16)b;
  return v.u;
}
// A-fragment (16x32 bf16) K-pair index for VGPR g, lane half
__device__ __forceinline__ int apair(int g, int half) {
  return half * 8 + g;
}

// ---------------- kernel 0: fill padded K/V with kv bias ----------------
__global__ __launch_bounds__(256) void init_kv_kernel(
    const float* __restrict__ kv_b,
    unsigned short* __restrict__ kpad, unsigned short* __restrict__ vpad) {
  size_t i = (size_t)blockIdx.x * 256 + threadIdx.x;
  if (i >= (size_t)NPAD * DIM) return;
  int c = (int)(i % DIM);
  kpad[i] = f2bf(kv_b[c]);
  vpad[i] = f2bf(kv_b[DIM + c]);
}

// ---------------- kernel 1: biasT[h][k][q] = bias_table[rpi[q,k]][h] ----
__global__ __launch_bounds__(256) void bias_kernel(
    const int* __restrict__ rpi, const float* __restrict__ table,
    unsigned short* __restrict__ biasT) {
  int k = blockIdx.x;       // 0..575
  int h = blockIdx.y;       // 0..5
  int q = threadIdx.x;      // 0..255
  int idx = rpi[q * NKV + k];
  biasT[((size_t)h * NKV + k) * NQ + q] = f2bf(table[(size_t)idx * HEADS + h]);
}

// ---------------- kernel 2: fused q/gate/kv GEMM ------------------------
// Y(51200 x 768) = x(51200 x 192) @ [q_w | gating_w | kv_w]
__global__ __launch_bounds__(256) void qkvg_kernel(
    const float* __restrict__ x,
    const float* __restrict__ q_w,  const float* __restrict__ q_b,
    const float* __restrict__ kv_w, const float* __restrict__ kv_b,
    const float* __restrict__ g_w,  const float* __restrict__ g_b,
    unsigned short* __restrict__ qb, unsigned short* __restrict__ gb,
    unsigned short* __restrict__ kpad, unsigned short* __restrict__ vpad) {
  __shared__ __align__(16) unsigned Al[128][16];   // 128 rows x 16 K-pairs
  __shared__ __align__(16) unsigned Bl[16][64];    // 16 K-pairs x 64 cols
  const int tid = threadIdx.x;
  const int wave = tid >> 5, lane = tid & 31, half = lane >> 4, ln = lane & 15;
  const int m0 = blockIdx.x * 128;
  const int n0 = blockIdx.y * 64;
  const float* Wsrc; int wcols, c0, reg;
  if (n0 < 192)      { Wsrc = q_w;  wcols = 192; c0 = n0;       reg = 0; }
  else if (n0 < 384) { Wsrc = g_w;  wcols = 192; c0 = n0 - 192; reg = 1; }
  else               { Wsrc = kv_w; wcols = 384; c0 = n0 - 384; reg = 2; }
  const int mw = wave & 3, nw_ = wave >> 2;
  v8f acc[2][2];
  const v8f vzero = {0.f,0.f,0.f,0.f,0.f,0.f,0.f,0.f};
  acc[0][0] = vzero; acc[0][1] = vzero; acc[1][0] = vzero; acc[1][1] = vzero;

  for (int kk = 0; kk < DIM; kk += 32) {
    { // A tile: 128 x 32 -> bf16 pairs
      int row = tid >> 1, cg = (tid & 1) * 16;
      const float* src = x + (size_t)(m0 + row) * DIM + kk + cg;
      #pragma unroll
      for (int i = 0; i < 8; ++i)
        Al[row][cg / 2 + i] = packbf2(src[2 * i], src[2 * i + 1]);
    }
    { // B tile: 32 x 64
      int k2 = tid & 15, cg = (tid >> 4) * 4;
      const float* r0 = Wsrc + (size_t)(kk + 2 * k2) * wcols + c0 + cg;
      const float* r1 = r0 + wcols;
      #pragma unroll
      for (int i = 0; i < 4; ++i) Bl[k2][cg + i] = packbf2(r0[i], r1[i]);
    }
    __syncthreads();
    Frag af[2], bfx[2];
    #pragma unroll
    for (int rt = 0; rt < 2; ++rt) {
      int row = mw * 32 + rt * 16 + ln;
      #pragma unroll
      for (int g = 0; g < 8; ++g) af[rt].u[g] = Al[row][apair(g, half)];
    }
    #pragma unroll
    for (int ct = 0; ct < 2; ++ct) {
      int col = nw_ * 32 + ct * 16 + ln;
      #pragma unroll
      for (int g = 0; g < 8; ++g) bfx[ct].u[g] = Bl[half * 8 + g][col];
    }
    #pragma unroll
    for (int rt = 0; rt < 2; ++rt)
      #pragma unroll
      for (int ct = 0; ct < 2; ++ct)
        acc[rt][ct] = __builtin_amdgcn_wmma_f32_16x16x32_bf16(
            false, af[rt].v, false, bfx[ct].v, (short)0, acc[rt][ct], false, false);
    __syncthreads();
  }
  // epilogue
  #pragma unroll
  for (int rt = 0; rt < 2; ++rt)
    #pragma unroll
    for (int ct = 0; ct < 2; ++ct)
      #pragma unroll
      for (int r = 0; r < 8; ++r) {
        int gm = m0 + mw * 32 + rt * 16 + half * 8 + r;
        int gn = n0 + nw_ * 32 + ct * 16 + ln;
        float val = acc[rt][ct][r];
        if (reg == 0) {
          qb[(size_t)gm * DIM + gn] = f2bf((val + q_b[gn]) * QSCALE);
        } else if (reg == 1) {
          int c = gn - 192;
          gb[(size_t)gm * DIM + c] = f2bf(val + g_b[c]);
        } else {
          int c = gn - 384;
          val += kv_b[c];
          int bt = gm / (IMGH * IMGW), rem = gm % (IMGH * IMGW);
          int y = rem / IMGW, xx = rem % IMGW;
          size_t pos = (((size_t)bt * PADH + y + 4) * PADH + xx + 4) * DIM;
          if (c < DIM) kpad[pos + c] = f2bf(val);
          else         vpad[pos + (c - DIM)] = f2bf(val);
        }
      }
}

// ---------------- kernel 3: fused flash attention per (window, head) ----
__global__ __launch_bounds__(256) void attn_kernel(
    const unsigned short* __restrict__ qb,
    const unsigned short* __restrict__ kpad,
    const unsigned short* __restrict__ vpad,
    const unsigned short* __restrict__ biasT,
    unsigned short* __restrict__ ao) {
  __shared__ __align__(16) unsigned Kl[64][16];            // key x d-pair
  __shared__ __align__(16) unsigned Vl[32][32];            // d   x key-pair
  __shared__ __align__(16) unsigned short Pl[8][32][32];   // per-wave P
  const int tid = threadIdx.x;
  const int wave = tid >> 5, lane = tid & 31, half = lane >> 4, ln = lane & 15;
  const int bid = blockIdx.x;
  const int head = bid % HEADS;
  const int win = bid / HEADS;
  const int bt = win / NWIN, widx = win % NWIN;
  const int wy = widx / NWIN1, wx = widx % NWIN1;

  // Q fragments (q already scaled)
  Frag qf[2];
  #pragma unroll
  for (int rt = 0; rt < 2; ++rt) {
    int qr = wave * 32 + rt * 16 + ln;
    int qy = qr >> 4, qx = qr & 15;
    size_t tok = ((size_t)bt * IMGH + wy * WSZ + qy) * IMGW + wx * WSZ + qx;
    const unsigned* src = (const unsigned*)(qb + tok * DIM + head * HDIM);
    #pragma unroll
    for (int g = 0; g < 8; ++g) qf[rt].u[g] = src[apair(g, half)];
  }

  const v8f vzero = {0.f,0.f,0.f,0.f,0.f,0.f,0.f,0.f};
  v8f acc[2][2]; acc[0][0]=vzero; acc[0][1]=vzero; acc[1][0]=vzero; acc[1][1]=vzero;
  float rmax[2][8], rsum[2][8];
  #pragma unroll
  for (int rt = 0; rt < 2; ++rt)
    #pragma unroll
    for (int r = 0; r < 8; ++r) { rmax[rt][r] = -3.0e38f; rsum[rt][r] = 0.f; }

  for (int c = 0; c < NKV / 64; ++c) {
    __syncthreads();
    { // K chunk: 64 keys x 32 dims
      int kc = tid >> 2, part = tid & 3;
      int kidx = c * 64 + kc;
      int ky = kidx / OWSZ, kx = kidx % OWSZ;
      size_t pos = ((size_t)bt * PADH + wy * WSZ + ky) * PADH + wx * WSZ + kx;
      const uint4* src = (const uint4*)(kpad + pos * DIM + head * HDIM);
      *(uint4*)&Kl[kc][part * 4] = src[part];
      // prefetch next chunk's K row (lowers to global_prefetch_b8 on gfx1250)
      if (c + 1 < NKV / 64) {
        int nk = kidx + 64;
        int nky = nk / OWSZ, nkx = nk % OWSZ;
        size_t npos = ((size_t)bt * PADH + wy * WSZ + nky) * PADH + wx * WSZ + nkx;
        __builtin_prefetch(kpad + npos * DIM + head * HDIM, 0, 0);
        __builtin_prefetch(vpad + npos * DIM + head * HDIM, 0, 0);
      }
    }
    { // V chunk transposed: Vl[d][keypair]
      int d = tid & 31, kp0 = (tid >> 5) * 4;
      #pragma unroll
      for (int i = 0; i < 4; ++i) {
        int kp = kp0 + i;
        int k0i = c * 64 + 2 * kp, k1i = k0i + 1;
        int ky0 = k0i / OWSZ, kx0 = k0i % OWSZ;
        int ky1 = k1i / OWSZ, kx1 = k1i % OWSZ;
        size_t p0 = ((size_t)bt * PADH + wy * WSZ + ky0) * PADH + wx * WSZ + kx0;
        size_t p1 = ((size_t)bt * PADH + wy * WSZ + ky1) * PADH + wx * WSZ + kx1;
        unsigned lo = vpad[p0 * DIM + head * HDIM + d];
        unsigned hi = vpad[p1 * DIM + head * HDIM + d];
        Vl[d][kp] = lo | (hi << 16);
      }
    }
    __syncthreads();

    #pragma unroll
    for (int p = 0; p < 2; ++p) {        // 32-key groups
      float sc[2][2][8];
      #pragma unroll
      for (int s = 0; s < 2; ++s) {      // 16-key tiles
        int kt = p * 2 + s;
        Frag bk;
        #pragma unroll
        for (int g = 0; g < 8; ++g) bk.u[g] = Kl[kt * 16 + ln][half * 8 + g];
        int kidx = c * 64 + kt * 16 + ln;
        int ky = kidx / OWSZ, kx = kidx % OWSZ;
        int py = wy * WSZ + ky, px = wx * WSZ + kx;
        float maskadd = (py >= 4 && py < IMGH + 4 && px >= 4 && px < IMGW + 4)
                            ? 0.f : -100.f;
        #pragma unroll
        for (int rt = 0; rt < 2; ++rt) {
          v8f sv = __builtin_amdgcn_wmma_f32_16x16x32_bf16(
              false, qf[rt].v, false, bk.v, (short)0, vzero, false, false);
          int qbase = wave * 32 + rt * 16 + half * 8;
          const uint4* bsrc =
              (const uint4*)(biasT + ((size_t)head * NKV + kidx) * NQ + qbase);
          uint4 bb = *bsrc;
          unsigned bu[4] = {bb.x, bb.y, bb.z, bb.w};
          #pragma unroll
          for (int j = 0; j < 4; ++j) {
            sc[rt][s][2*j]   = sv[2*j]   + bf2f((unsigned short)(bu[j] & 0xffff)) + maskadd;
            sc[rt][s][2*j+1] = sv[2*j+1] + bf2f((unsigned short)(bu[j] >> 16))    + maskadd;
          }
        }
      }
      // online softmax over this 32-key group
      #pragma unroll
      for (int rt = 0; rt < 2; ++rt) {
        float corr[8];
        #pragma unroll
        for (int r = 0; r < 8; ++r) {
          float t = fmaxf(sc[rt][0][r], sc[rt][1][r]);
          #pragma unroll
          for (int off = 1; off < 16; off <<= 1)
            t = fmaxf(t, __shfl_xor(t, off, 32));
          float mo = rmax[rt][r];
          float mn = fmaxf(mo, t);
          rmax[rt][r] = mn;
          corr[r] = __expf(mo - mn);
          float p0 = __expf(sc[rt][0][r] - mn);
          float p1 = __expf(sc[rt][1][r] - mn);
          float ps = p0 + p1;
          #pragma unroll
          for (int off = 1; off < 16; off <<= 1)
            ps += __shfl_xor(ps, off, 32);
          rsum[rt][r] = rsum[rt][r] * corr[r] + ps;
          Pl[wave][rt * 16 + half * 8 + r][ln]      = f2bf(p0);
          Pl[wave][rt * 16 + half * 8 + r][16 + ln] = f2bf(p1);
        }
        #pragma unroll
        for (int dt = 0; dt < 2; ++dt)
          #pragma unroll
          for (int r = 0; r < 8; ++r) acc[rt][dt][r] *= corr[r];
      }
      // O += P(32q x 32k) @ V(32k x 32d)
      Frag pa[2], bv[2];
      #pragma unroll
      for (int rt = 0; rt < 2; ++rt) {
        const unsigned* pr = (const unsigned*)&Pl[wave][rt * 16 + ln][0];
        #pragma unroll
        for (int g = 0; g < 8; ++g) pa[rt].u[g] = pr[apair(g, half)];
      }
      #pragma unroll
      for (int dt = 0; dt < 2; ++dt)
        #pragma unroll
        for (int g = 0; g < 8; ++g)
          bv[dt].u[g] = Vl[dt * 16 + ln][p * 16 + half * 8 + g];
      #pragma unroll
      for (int rt = 0; rt < 2; ++rt)
        #pragma unroll
        for (int dt = 0; dt < 2; ++dt)
          acc[rt][dt] = __builtin_amdgcn_wmma_f32_16x16x32_bf16(
              false, pa[rt].v, false, bv[dt].v, (short)0, acc[rt][dt], false, false);
    }
  }
  // normalize + store (gating applied in proj kernel)
  #pragma unroll
  for (int rt = 0; rt < 2; ++rt)
    #pragma unroll
    for (int dt = 0; dt < 2; ++dt)
      #pragma unroll
      for (int r = 0; r < 8; ++r) {
        int qr = wave * 32 + rt * 16 + half * 8 + r;
        int qy = qr >> 4, qx = qr & 15;
        size_t tok = ((size_t)bt * IMGH + wy * WSZ + qy) * IMGW + wx * WSZ + qx;
        float v = acc[rt][dt][r] / rsum[rt][r];
        ao[tok * DIM + head * HDIM + dt * 16 + ln] = f2bf(v);
      }
}

// ---------------- kernel 4: out = (ao * gates) @ proj_w + proj_b --------
__global__ __launch_bounds__(256) void proj_kernel(
    const unsigned short* __restrict__ ao, const unsigned short* __restrict__ gb,
    const float* __restrict__ pw, const float* __restrict__ pb,
    float* __restrict__ out) {
  __shared__ __align__(16) unsigned Al[128][16];
  __shared__ __align__(16) unsigned Bl[16][64];
  const int tid = threadIdx.x;
  const int wave = tid >> 5, lane = tid & 31, half = lane >> 4, ln = lane & 15;
  const int m0 = blockIdx.x * 128;
  const int n0 = blockIdx.y * 64;
  const int mw = wave & 3, nw_ = wave >> 2;
  const v8f vzero = {0.f,0.f,0.f,0.f,0.f,0.f,0.f,0.f};
  v8f acc[2][2]; acc[0][0]=vzero; acc[0][1]=vzero; acc[1][0]=vzero; acc[1][1]=vzero;

  for (int kk = 0; kk < DIM; kk += 32) {
    {
      int row = tid >> 1, cg = (tid & 1) * 16;
      const unsigned* sa = (const unsigned*)(ao + (size_t)(m0 + row) * DIM + kk + cg);
      const unsigned* sg = (const unsigned*)(gb + (size_t)(m0 + row) * DIM + kk + cg);
      #pragma unroll
      for (int i = 0; i < 8; ++i) {
        unsigned ua = sa[i], ug = sg[i];
        float lo = bf2f((unsigned short)(ua & 0xffff)) * bf2f((unsigned short)(ug & 0xffff));
        float hi = bf2f((unsigned short)(ua >> 16)) * bf2f((unsigned short)(ug >> 16));
        Al[row][cg / 2 + i] = packbf2(lo, hi);
      }
    }
    {
      int k2 = tid & 15, cg = (tid >> 4) * 4;
      const float* r0 = pw + (size_t)(kk + 2 * k2) * DIM + n0 + cg;
      const float* r1 = r0 + DIM;
      #pragma unroll
      for (int i = 0; i < 4; ++i) Bl[k2][cg + i] = packbf2(r0[i], r1[i]);
    }
    __syncthreads();
    Frag af[2], bfx[2];
    #pragma unroll
    for (int rt = 0; rt < 2; ++rt) {
      int row = mw * 32 + rt * 16 + ln;
      #pragma unroll
      for (int g = 0; g < 8; ++g) af[rt].u[g] = Al[row][apair(g, half)];
    }
    #pragma unroll
    for (int ct = 0; ct < 2; ++ct) {
      int col = nw_ * 32 + ct * 16 + ln;
      #pragma unroll
      for (int g = 0; g < 8; ++g) bfx[ct].u[g] = Bl[half * 8 + g][col];
    }
    #pragma unroll
    for (int rt = 0; rt < 2; ++rt)
      #pragma unroll
      for (int ct = 0; ct < 2; ++ct)
        acc[rt][ct] = __builtin_amdgcn_wmma_f32_16x16x32_bf16(
            false, af[rt].v, false, bfx[ct].v, (short)0, acc[rt][ct], false, false);
    __syncthreads();
  }
  #pragma unroll
  for (int rt = 0; rt < 2; ++rt)
    #pragma unroll
    for (int ct = 0; ct < 2; ++ct)
      #pragma unroll
      for (int r = 0; r < 8; ++r) {
        int gm = m0 + mw * 32 + rt * 16 + half * 8 + r;
        int gn = n0 + nw_ * 32 + ct * 16 + ln;
        out[(size_t)gm * DIM + gn] = acc[rt][ct][r] + pb[gn];
      }
}

extern "C" void kernel_launch(void* const* d_in, const int* in_sizes, int n_in,
                              void* d_out, int out_size, void* d_ws, size_t ws_size,
                              hipStream_t stream) {
  const float* x    = (const float*)d_in[0];
  const int*   rpi  = (const int*)d_in[1];
  // d_in[2] attn_mask: recomputed geometrically (shift_size==0), unused
  const float* q_w  = (const float*)d_in[3];
  const float* q_b  = (const float*)d_in[4];
  const float* kv_w = (const float*)d_in[5];
  const float* kv_b = (const float*)d_in[6];
  const float* btab = (const float*)d_in[7];
  const float* g_w  = (const float*)d_in[8];
  const float* g_b  = (const float*)d_in[9];
  const float* p_w  = (const float*)d_in[10];
  const float* p_b  = (const float*)d_in[11];

  unsigned short* qb    = (unsigned short*)d_ws;
  unsigned short* gb    = qb    + (size_t)NTOK * DIM;
  unsigned short* kpad  = gb    + (size_t)NTOK * DIM;
  unsigned short* vpad  = kpad  + (size_t)NPAD * DIM;
  unsigned short* ao    = vpad  + (size_t)NPAD * DIM;
  unsigned short* biasT = ao    + (size_t)NTOK * DIM;

  init_kv_kernel<<<((size_t)NPAD * DIM + 255) / 256, 256, 0, stream>>>(kv_b, kpad, vpad);
  bias_kernel<<<dim3(NKV, HEADS), NQ, 0, stream>>>(rpi, btab, biasT);
  qkvg_kernel<<<dim3(NTOK / 128, 12), 256, 0, stream>>>(
      x, q_w, q_b, kv_w, kv_b, g_w, g_b, qb, gb, kpad, vpad);
  attn_kernel<<<BATCH * NWIN * HEADS, 256, 0, stream>>>(qb, kpad, vpad, biasT, ao);
  proj_kernel<<<dim3(NTOK / 128, 3), 256, 0, stream>>>(ao, gb, p_w, p_b, (float*)d_out);
}
